// CausalSelfAttention_3564822855833
// MI455X (gfx1250) — matrix operand order repaired and checked by reading, not compiled, not run on hardware
//
#include <hip/hip_runtime.h>
#include <hip/hip_bf16.h>

// ---------------------------------------------------------------------------
// CDNA5 / gfx1250 causal self-attention, bf16 WMMA pipeline.
// ---------------------------------------------------------------------------

typedef __attribute__((ext_vector_type(16))) __bf16 v16bf;
typedef __attribute__((ext_vector_type(4)))  __bf16 v4bf;
typedef __attribute__((ext_vector_type(8)))  float  v8f;

#define B_SZ   2
#define T_SZ   2048
#define C_SZ   1024
#define H_SZ   16
#define DH_SZ  64
#define M_SZ   (B_SZ * T_SZ)   // 4096 rows of x

union Frag {
    v16bf v;
    uint4 q[2];
};

__device__ __forceinline__ v8f wmma_bf16(v16bf a, v16bf b, v8f c) {
    // (neg_a, A, neg_b, B, c_mod, C, reuse_a, reuse_b)
    return __builtin_amdgcn_wmma_f32_16x16x32_bf16(false, a, false, b, (short)0, c,
                                                   false, false);
}

__device__ __forceinline__ float rmax16(float v) {
    v = fmaxf(v, __shfl_xor(v, 1, 16));
    v = fmaxf(v, __shfl_xor(v, 2, 16));
    v = fmaxf(v, __shfl_xor(v, 4, 16));
    v = fmaxf(v, __shfl_xor(v, 8, 16));
    return v;
}
__device__ __forceinline__ float rsum16(float v) {
    v += __shfl_xor(v, 1, 16);
    v += __shfl_xor(v, 2, 16);
    v += __shfl_xor(v, 4, 16);
    v += __shfl_xor(v, 8, 16);
    return v;
}

// ---------------------------------------------------------------------------
// fp32 -> bf16 conversion, 4 elements per lane
// ---------------------------------------------------------------------------
__global__ void cvt_f32_to_bf16(const float* __restrict__ in,
                                __bf16* __restrict__ out, int n4) {
    int i = blockIdx.x * blockDim.x + threadIdx.x;
    if (i < n4) {
        float4 f = ((const float4*)in)[i];
        v4bf o;
        o[0] = (__bf16)f.x; o[1] = (__bf16)f.y;
        o[2] = (__bf16)f.z; o[3] = (__bf16)f.w;
        ((v4bf*)out)[i] = o;
    }
}

// ---------------------------------------------------------------------------
// Y = A(MxK) @ W(NxK)^T, bf16 inputs, per-wave 32(M) x 64(N) tile,
// register double-buffered so loads for k+32 overlap WMMAs for k.
// mode 0: fp32 row-major to outF
// mode 1: bf16 to outH in [B,H,T,Dh] layout (for Q, K)
// mode 2: bf16 to outH in [B,H,Dh,T] layout (transposed V)
// ---------------------------------------------------------------------------
__device__ __forceinline__ void load_ab(Frag* a, Frag* b,
                                        const __bf16* const* arow,
                                        const __bf16* const* wrow, int k0) {
#pragma unroll
    for (int s = 0; s < 2; ++s) {
        a[s].q[0] = *(const uint4*)(arow[s] + k0);
        a[s].q[1] = *(const uint4*)(arow[s] + k0 + 8);
    }
#pragma unroll
    for (int c = 0; c < 4; ++c) {
        b[c].q[0] = *(const uint4*)(wrow[c] + k0);
        b[c].q[1] = *(const uint4*)(wrow[c] + k0 + 8);
    }
}

__global__ void gemm_bf16_xWt(const __bf16* __restrict__ A,
                              const __bf16* __restrict__ W,
                              int M, int N, int K,
                              float* __restrict__ outF,
                              __bf16* __restrict__ outH, int mode) {
    const int lane = threadIdx.x & 31;
    const int wave = threadIdx.x >> 5;
    const int gw   = blockIdx.x * (blockDim.x >> 5) + wave;
    const int mtiles = M >> 5;          // 32-row tiles
    const int mt = gw % mtiles;
    const int nt = gw / mtiles;
    const int M0 = mt << 5;
    const int N0 = nt << 6;
    const int hf = lane >> 4;           // which 16-lane half
    const int lo = lane & 15;

    v8f acc[8] = {v8f{}, v8f{}, v8f{}, v8f{}, v8f{}, v8f{}, v8f{}, v8f{}};

    // A-fragment lane bases: rows M0+lo and M0+16+lo, K chunks {hf*16, hf*16+8}
    const __bf16* arow[2];
    arow[0] = A + (size_t)(M0 + lo) * K + hf * 16;
    arow[1] = arow[0] + (size_t)16 * K;
    // B-fragment lane bases: column N0+c*16+lo = row of W, contiguous 32B
    const __bf16* wrow[4];
#pragma unroll
    for (int c = 0; c < 4; ++c)
        wrow[c] = W + (size_t)(N0 + c * 16 + lo) * K + hf * 16;

    Frag ca[2], cb[4];
    load_ab(ca, cb, arow, wrow, 0);

    for (int k0 = 0; k0 < K; k0 += 32) {
        Frag na[2], nb[4];
        if (k0 + 32 < K)
            load_ab(na, nb, arow, wrow, k0 + 32);   // in flight during WMMAs
#pragma unroll
        for (int s = 0; s < 2; ++s)
#pragma unroll
            for (int c = 0; c < 4; ++c)
                acc[s * 4 + c] = wmma_bf16(ca[s].v, cb[c].v, acc[s * 4 + c]);
#pragma unroll
        for (int s = 0; s < 2; ++s) ca[s] = na[s];
#pragma unroll
        for (int c = 0; c < 4; ++c) cb[c] = nb[c];
    }

#pragma unroll
    for (int s = 0; s < 2; ++s) {
#pragma unroll
        for (int c = 0; c < 4; ++c) {
#pragma unroll
            for (int r = 0; r < 8; ++r) {
                float val = acc[s * 4 + c][r];
                int row = M0 + s * 16 + r + hf * 8;  // C/D layout: VGPR r -> M=r (+8)
                int col = N0 + c * 16 + lo;          // lane -> N
                if (mode == 0) {
                    outF[(size_t)row * N + col] = val;
                } else {
                    int t = row & (T_SZ - 1);
                    int b = row >> 11;
                    int h = col >> 6;
                    int d = col & (DH_SZ - 1);
                    size_t idx = (mode == 1)
                        ? (((size_t)(b * H_SZ + h) * T_SZ + t) * DH_SZ + d)
                        : (((size_t)(b * H_SZ + h) * DH_SZ + d) * T_SZ + t);
                    outH[idx] = (__bf16)val;
                }
            }
        }
    }
}

// ---------------------------------------------------------------------------
// Flash attention: one wave owns 16 query rows of one (b,h); streams 32 keys
// per iteration.  Q,K in [B,H,T,Dh] bf16, V transposed [B,H,Dh,T] bf16.
// Output Y in [B,T,H*Dh] bf16 (row-major activations for the final GEMM).
// ---------------------------------------------------------------------------
__global__ void flash_attn(const __bf16* __restrict__ Qm,
                           const __bf16* __restrict__ Km,
                           const __bf16* __restrict__ Vt,
                           __bf16* __restrict__ Y) {
    __shared__ __bf16 ldsTile[4 * 16 * 32];   // 1 KB per wave
    const int lane = threadIdx.x & 31;
    const int wave = threadIdx.x >> 5;
    const int gw   = blockIdx.x * 4 + wave;
    const int qt   = gw & 127;                // q tile within head
    const int bh   = gw >> 7;                 // batch*head
    const int r0   = qt << 4;                 // first query row of this wave
    const int hf   = lane >> 4;
    const int lo   = lane & 15;
    __bf16* myLds  = ldsTile + wave * (16 * 32);

    const size_t headQK = (size_t)bh * T_SZ * DH_SZ;   // base of this head
    const size_t headV  = (size_t)bh * DH_SZ * T_SZ;

    // Q fragments (A layout), full Dh = 64 -> two K=32 fragments
    const __bf16* qrow = Qm + headQK + (size_t)(r0 + lo) * DH_SZ + hf * 16;
    Frag aq0, aq1;
    aq0.q[0] = *(const uint4*)(qrow);
    aq0.q[1] = *(const uint4*)(qrow + 8);
    aq1.q[0] = *(const uint4*)(qrow + 32);
    aq1.q[1] = *(const uint4*)(qrow + 40);

    float rowmax[8], rowsum[8];
#pragma unroll
    for (int r = 0; r < 8; ++r) { rowmax[r] = -__builtin_inff(); rowsum[r] = 0.f; }
    v8f o[4] = {v8f{}, v8f{}, v8f{}, v8f{}};

    const int nblk = ((r0 + 15) >> 5) + 1;    // causal: keys <= r0+15
    for (int kb = 0; kb < nblk; ++kb) {
        const int key0 = kb << 5;

        // ---- S = Q @ K^T for 32 keys (two 16-key fragments) ----
        v8f s[2] = {v8f{}, v8f{}};
#pragma unroll
        for (int f = 0; f < 2; ++f) {
            const __bf16* krow = Km + headQK + (size_t)(key0 + f * 16 + lo) * DH_SZ;
            Frag b0, b1;
            b0.q[0] = *(const uint4*)(krow + hf * 16);
            b0.q[1] = *(const uint4*)(krow + hf * 16 + 8);
            b1.q[0] = *(const uint4*)(krow + 32 + hf * 16);
            b1.q[1] = *(const uint4*)(krow + 32 + hf * 16 + 8);
            s[f] = wmma_bf16(aq0.v, b0.v, s[f]);
            s[f] = wmma_bf16(aq1.v, b1.v, s[f]);
        }

        // ---- V fragments: issue loads now so they overlap the softmax VALU ----
        Frag bv[4];
#pragma unroll
        for (int c = 0; c < 4; ++c) {
            const __bf16* vrow = Vt + headV + (size_t)(c * 16 + lo) * T_SZ
                               + key0 + hf * 16;
            bv[c].q[0] = *(const uint4*)(vrow);
            bv[c].q[1] = *(const uint4*)(vrow + 8);
        }
        // speculative prefetch of next key block's K rows (global_prefetch_b8)
        if (kb + 1 < nblk)
            __builtin_prefetch(Km + headQK + (size_t)(key0 + 32 + lane) * DH_SZ, 0, 1);

        // ---- scale + causal mask ----
#pragma unroll
        for (int f = 0; f < 2; ++f) {
#pragma unroll
            for (int r = 0; r < 8; ++r) {
                int row = r0 + r + hf * 8;
                int col = key0 + f * 16 + lo;
                float v = s[f][r] * 0.125f;   // 1/sqrt(64)
                s[f][r] = (col <= row) ? v : -__builtin_inff();
            }
        }

        // ---- online softmax statistics (rows live in 16-lane halves) ----
        float alpha[8];
#pragma unroll
        for (int r = 0; r < 8; ++r) {
            float bm = rmax16(fmaxf(s[0][r], s[1][r]));
            float mn = fmaxf(rowmax[r], bm);
            alpha[r] = __expf(rowmax[r] - mn);
            rowmax[r] = mn;
        }
#pragma unroll
        for (int f = 0; f < 2; ++f)
#pragma unroll
            for (int r = 0; r < 8; ++r)
                s[f][r] = __expf(s[f][r] - rowmax[r]);
#pragma unroll
        for (int r = 0; r < 8; ++r) {
            float bs = rsum16(s[0][r] + s[1][r]);
            rowsum[r] = rowsum[r] * alpha[r] + bs;
        }
#pragma unroll
        for (int c = 0; c < 4; ++c)
#pragma unroll
            for (int r = 0; r < 8; ++r)
                o[c][r] *= alpha[r];

        // ---- P: C-layout -> A-layout via per-wave LDS tile (in-order DS) ----
#pragma unroll
        for (int f = 0; f < 2; ++f)
#pragma unroll
            for (int r = 0; r < 8; ++r)
                myLds[(r + hf * 8) * 32 + f * 16 + lo] = (__bf16)s[f][r];
        __builtin_amdgcn_fence(__ATOMIC_SEQ_CST, "wavefront");
        Frag pa;
        pa.q[0] = *(const uint4*)(myLds + lo * 32 + hf * 16);
        pa.q[1] = *(const uint4*)(myLds + lo * 32 + hf * 16 + 8);
        __builtin_amdgcn_fence(__ATOMIC_SEQ_CST, "wavefront");

        // ---- O += P @ V ----
#pragma unroll
        for (int c = 0; c < 4; ++c)
            o[c] = wmma_bf16(pa.v, bv[c].v, o[c]);
    }

    // ---- normalize and write Y [B,T,C] bf16 ----
    const int b = bh >> 4;
    const int h = bh & 15;
#pragma unroll
    for (int c = 0; c < 4; ++c) {
#pragma unroll
        for (int r = 0; r < 8; ++r) {
            int t = r0 + r + hf * 8;
            float val = o[c][r] / rowsum[r];
            size_t idx = ((size_t)(b * T_SZ + t)) * C_SZ + h * DH_SZ + c * 16 + lo;
            Y[idx] = (__bf16)val;
        }
    }
}

// ---------------------------------------------------------------------------
// Host-side orchestration
// ---------------------------------------------------------------------------
extern "C" void kernel_launch(void* const* d_in, const int* in_sizes, int n_in,
                              void* d_out, int out_size, void* d_ws, size_t ws_size,
                              hipStream_t stream) {
    const float* x  = (const float*)d_in[0];
    const float* Wq = (const float*)d_in[1];
    const float* Wk = (const float*)d_in[2];
    const float* Wv = (const float*)d_in[3];
    const float* Wo = (const float*)d_in[4];

    __bf16* w   = (__bf16*)d_ws;
    __bf16* Xh  = w;                                   // 4096*1024
    __bf16* Whq = Xh  + (size_t)M_SZ * C_SZ;           // 1024*1024 each
    __bf16* Whk = Whq + (size_t)C_SZ * C_SZ;
    __bf16* Whv = Whk + (size_t)C_SZ * C_SZ;
    __bf16* Who = Whv + (size_t)C_SZ * C_SZ;
    __bf16* Qb  = Who + (size_t)C_SZ * C_SZ;           // [B,H,T,Dh]
    __bf16* Kb  = Qb  + (size_t)M_SZ * C_SZ;
    __bf16* Vtb = Kb  + (size_t)M_SZ * C_SZ;           // [B,H,Dh,T]
    __bf16* Yb  = Vtb + (size_t)M_SZ * C_SZ;           // [B,T,C]

    // 1. convert activations + weights to bf16
    {
        int n4 = (M_SZ * C_SZ) / 4;
        cvt_f32_to_bf16<<<n4 / 256, 256, 0, stream>>>(x, Xh, n4);
        int w4 = (C_SZ * C_SZ) / 4;
        cvt_f32_to_bf16<<<w4 / 256, 256, 0, stream>>>(Wq, Whq, w4);
        cvt_f32_to_bf16<<<w4 / 256, 256, 0, stream>>>(Wk, Whk, w4);
        cvt_f32_to_bf16<<<w4 / 256, 256, 0, stream>>>(Wv, Whv, w4);
        cvt_f32_to_bf16<<<w4 / 256, 256, 0, stream>>>(Wo, Who, w4);
    }

    // 2. Q/K/V projections: waves = (M/32)*(N/64) = 2048 -> 256 blocks x 8 waves
    const int gemmBlocks = (M_SZ / 32) * (C_SZ / 64) / 8;
    gemm_bf16_xWt<<<gemmBlocks, 256, 0, stream>>>(Xh, Whq, M_SZ, C_SZ, C_SZ,
                                                  nullptr, Qb, 1);
    gemm_bf16_xWt<<<gemmBlocks, 256, 0, stream>>>(Xh, Whk, M_SZ, C_SZ, C_SZ,
                                                  nullptr, Kb, 1);
    gemm_bf16_xWt<<<gemmBlocks, 256, 0, stream>>>(Xh, Whv, M_SZ, C_SZ, C_SZ,
                                                  nullptr, Vtb, 2);

    // 3. flash attention: B*H*(T/16) = 4096 waves -> 1024 blocks x 4 waves
    flash_attn<<<1024, 128, 0, stream>>>(Qb, Kb, Vtb, Yb);

    // 4. output projection -> fp32 d_out
    gemm_bf16_xWt<<<gemmBlocks, 256, 0, stream>>>(Yb, Who, M_SZ, C_SZ, C_SZ,
                                                  (float*)d_out, nullptr, 0);
}
